// ScalableNeRFModel_47536698032162
// MI455X (gfx1250) — matrix operand order repaired and checked by reading, not compiled, not run on hardware
//
#include <hip/hip_runtime.h>
#include <hip/hip_bf16.h>
#include <math.h>

// ---------------------------------------------------------------------------
// ScalableNeRF on MI455X (gfx1250): hash-grid encode + 8-layer MLP via WMMA.
//
//   kernel 1 (prep): transpose+convert w0 (91x256, K-padded to 96) and
//                    wh (7x256x256) weights from f32 to f16 into d_ws,
//                    layout Wt[n][k] so B-fragments are contiguous 32B loads.
//   kernel 2 (main): per block of 128 rows (128 threads = 4 waves,
//                    2 M-tiles per wave):
//     - hash-grid gather (8 levels x 8 feats) + view sin/cos encode -> LDS f16
//     - 8 MLP layers via v_wmma_f32_16x16x32_f16, activations ping-pong in LDS
//     - density / sigmoid(color) heads, scalar per-thread.
//   LDS = 2 x 128 x 264 x 2B = 135168 B/block -> 2 blocks co-resident per WGP
//   (270 KB <= 320 KB), so one block's WMMA phase overlaps the other block's
//   gather/encode/head phases instead of idling the matrix pipes at barriers.
// ---------------------------------------------------------------------------

#define NLVL      8
#define FEAT      8
#define WIDTH     256
#define THREADS   128        // threads per block (4 waves)
#define ROWS      128        // rows (points) per workgroup
#define LDSTRIDE  264        // f16 row stride in LDS (bank-skew padding)
#define KPAD0     96         // layer-0 K (91 padded to multiple of 32)

typedef _Float16 half_t;
typedef __attribute__((ext_vector_type(16))) _Float16 v16h;
typedef __attribute__((ext_vector_type(8)))  _Float16 v8h;
typedef __attribute__((ext_vector_type(8)))  float    v8f;
typedef __attribute__((ext_vector_type(4)))  float    v4f;

// ----------------------------- weight prep ---------------------------------
// Wt0: [256][96] f16  (Wt0[n][k] = w0[k][n], zero pad k>=91)       24576 halfs
// WtH: 7 x [256][256] f16 (WtH[l][n][k] = wh[l][k][n])            458752 halfs
#define WT0_ELEMS (WIDTH * KPAD0)
#define WTH_ELEMS (7 * WIDTH * WIDTH)

__global__ __launch_bounds__(256)
void nerf_prep_weights(const float* __restrict__ w0,
                       const float* __restrict__ wh,
                       half_t* __restrict__ wt) {
  int tid = blockIdx.x * 256 + threadIdx.x;
  if (tid < WT0_ELEMS) {
    int n = tid / KPAD0, k = tid % KPAD0;
    wt[tid] = (k < 91) ? (half_t)w0[k * WIDTH + n] : (half_t)0.0f;
  } else {
    int t2 = tid - WT0_ELEMS;
    if (t2 < WTH_ELEMS) {
      int l = t2 / (WIDTH * WIDTH);
      int r = t2 % (WIDTH * WIDTH);
      int n = r / WIDTH, k = r % WIDTH;
      wt[WT0_ELEMS + t2] = (half_t)wh[l * WIDTH * WIDTH + k * WIDTH + n];
    }
  }
}

// ------------------------------ MLP layer ----------------------------------
// 4 waves per block, wave w owns M-tiles {2w, 2w+1} (32 rows): every
// B fragment is reused by 2 WMMAs.
// B fragment (32x16 f16, this code's contiguous-k order): lane L = column N=L%16,
//   K range = (L<16 ? 0..15 : 16..31) -> 16 contiguous halfs of Wt[n][k].
// A fragment (16x32 f16): lane L = row M=L%16,
//   K = kh*16..kh*16+15 contiguous (kh = L>>4), matching B.
// D (f32 16x16): VGPR j -> row M = j + 8*kh, col N = L%16.
template <int KSTEPS, bool RELU>
__device__ __forceinline__
void mlp_layer(const half_t* __restrict__ Wt, int kstride,
               const float* __restrict__ bias,
               const half_t* __restrict__ in,   // LDS, stride LDSTRIDE
               half_t* __restrict__ out) {      // LDS, stride LDSTRIDE
  const int lane = threadIdx.x & 31;
  const int wv   = threadIdx.x >> 5;
  const int nl   = lane & 15;
  const int kh   = lane >> 4;
  const int m0   = wv * 2;
  const int m1   = wv * 2 + 1;

  const half_t* a0base = in + (m0 * 16 + nl) * LDSTRIDE;
  const half_t* a1base = in + (m1 * 16 + nl) * LDSTRIDE;

  for (int nt = 0; nt < 16; ++nt) {
    v8f acc0 = {};
    v8f acc1 = {};
    const half_t* bbase = Wt + (nt * 16 + nl) * kstride + kh * 16;
#pragma unroll
    for (int ks = 0; ks < KSTEPS; ++ks) {
      v16h bf = *(const v16h*)(bbase + ks * 32);
      union { v16h v; v8h h[2]; } a0, a1;
      a0.h[0] = *(const v8h*)(a0base + ks * 32 + kh * 16);
      a0.h[1] = *(const v8h*)(a0base + ks * 32 + kh * 16 + 8);
      a1.h[0] = *(const v8h*)(a1base + ks * 32 + kh * 16);
      a1.h[1] = *(const v8h*)(a1base + ks * 32 + kh * 16 + 8);
      acc0 = __builtin_amdgcn_wmma_f32_16x16x32_f16(
          false, a0.v, false, bf, (short)0, acc0, false, false);
      acc1 = __builtin_amdgcn_wmma_f32_16x16x32_f16(
          false, a1.v, false, bf, (short)0, acc1, false, false);
    }
    const float bv = bias[nt * 16 + nl];
    const int col = nt * 16 + nl;
#pragma unroll
    for (int j = 0; j < 8; ++j) {
      float v0 = acc0[j] + bv;
      float v1 = acc1[j] + bv;
      if (RELU) { v0 = fmaxf(v0, 0.0f); v1 = fmaxf(v1, 0.0f); }
      int r0 = m0 * 16 + j + kh * 8;
      int r1 = m1 * 16 + j + kh * 8;
      out[r0 * LDSTRIDE + col] = (half_t)v0;
      out[r1 * LDSTRIDE + col] = (half_t)v1;
    }
  }
}

// ------------------------------ main kernel --------------------------------
struct NerfParams {
  const float* pos;
  const float* view;
  const float* tab[NLVL];
  const float* b0;
  const float* bh;     // [7][256]
  const float* wd;     // [256][1]
  const float* bd;     // [1]
  const float* wc;     // [256][3]
  const float* bc;     // [3]
  const half_t* wt0;   // [256][96]   f16 transposed
  const half_t* wth;   // 7x[256][256] f16 transposed
  float* out;          // density [N] then color [N][3]
  int n;
};

__global__ __launch_bounds__(THREADS)
void nerf_main(NerfParams P) {
  __shared__ half_t bufA[ROWS * LDSTRIDE];
  __shared__ half_t bufB[ROWS * LDSTRIDE];

  const int t = threadIdx.x;
  const int g = blockIdx.x * ROWS + t;

  // ---------------- encoding: one thread = one point -----------------------
  {
    const float px = P.pos[g * 3 + 0];
    const float py = P.pos[g * 3 + 1];
    const float pz = P.pos[g * 3 + 2];
    const int RES[NLVL] = {32, 64, 128, 256, 512, 1024, 2048, 2048};
    const int TSZ[NLVL] = {32768, 262144, 524288, 524288,
                           524288, 524288, 524288, 524288};
    half_t* row = bufA + t * LDSTRIDE;
#pragma unroll
    for (int l = 0; l < NLVL; ++l) {
      const float rf = (float)RES[l];
      const float hx = fminf(fmaxf((px + 1.0f) * 0.5f * rf, 0.0f), rf - 1.0f);
      const float hy = fminf(fmaxf((py + 1.0f) * 0.5f * rf, 0.0f), rf - 1.0f);
      const float hz = fminf(fmaxf((pz + 1.0f) * 0.5f * rf, 0.0f), rf - 1.0f);
      float idxf = hx * (rf * rf) + hy * rf + hz;
      idxf = fminf(fmaxf(idxf, 0.0f), (float)(TSZ[l] - 1));
      const int idx = (int)idxf;
      const v4f f0 = *(const v4f*)(P.tab[l] + (size_t)idx * FEAT);
      const v4f f1 = *(const v4f*)(P.tab[l] + (size_t)idx * FEAT + 4);
#pragma unroll
      for (int j = 0; j < 4; ++j) {
        row[l * FEAT + j]     = (half_t)f0[j];
        row[l * FEAT + 4 + j] = (half_t)f1[j];
      }
    }
    const float PI = 3.14159265358979323846f;
#pragma unroll
    for (int i = 0; i < 3; ++i) {
      const float v = P.view[g * 3 + i];
      row[64 + i] = (half_t)v;
#pragma unroll
      for (int f = 0; f < 4; ++f) {
        const float a = (float)(1 << f) * PI * v;
        row[67 + f * 6 + i] = (half_t)__sinf(a);
        row[70 + f * 6 + i] = (half_t)__cosf(a);
      }
    }
#pragma unroll
    for (int c = 91; c < KPAD0; ++c) row[c] = (half_t)0.0f;
  }
  __syncthreads();

  // ---------------- MLP: 1 input layer + 6 ReLU + 1 final ------------------
  mlp_layer<3, true>(P.wt0, KPAD0, P.b0, bufA, bufB);
  __syncthreads();
  mlp_layer<8, true>(P.wth + 0 * WIDTH * WIDTH, WIDTH, P.bh + 0 * WIDTH, bufB, bufA);
  __syncthreads();
  mlp_layer<8, true>(P.wth + 1 * WIDTH * WIDTH, WIDTH, P.bh + 1 * WIDTH, bufA, bufB);
  __syncthreads();
  mlp_layer<8, true>(P.wth + 2 * WIDTH * WIDTH, WIDTH, P.bh + 2 * WIDTH, bufB, bufA);
  __syncthreads();
  mlp_layer<8, true>(P.wth + 3 * WIDTH * WIDTH, WIDTH, P.bh + 3 * WIDTH, bufA, bufB);
  __syncthreads();
  mlp_layer<8, true>(P.wth + 4 * WIDTH * WIDTH, WIDTH, P.bh + 4 * WIDTH, bufB, bufA);
  __syncthreads();
  mlp_layer<8, true>(P.wth + 5 * WIDTH * WIDTH, WIDTH, P.bh + 5 * WIDTH, bufA, bufB);
  __syncthreads();
  mlp_layer<8, false>(P.wth + 6 * WIDTH * WIDTH, WIDTH, P.bh + 6 * WIDTH, bufB, bufA);
  __syncthreads();

  // ---------------- heads: density (256->1), color (256->3) ---------------
  {
    const half_t* x = bufA + t * LDSTRIDE;
    float ad = 0.0f, ac0 = 0.0f, ac1 = 0.0f, ac2 = 0.0f;
#pragma unroll 8
    for (int k = 0; k < WIDTH; ++k) {
      const float xv = (float)x[k];
      ad  += xv * P.wd[k];
      ac0 += xv * P.wc[k * 3 + 0];
      ac1 += xv * P.wc[k * 3 + 1];
      ac2 += xv * P.wc[k * 3 + 2];
    }
    const float density = ad + P.bd[0];
    const float c0 = 1.0f / (1.0f + __expf(-(ac0 + P.bc[0])));
    const float c1 = 1.0f / (1.0f + __expf(-(ac1 + P.bc[1])));
    const float c2 = 1.0f / (1.0f + __expf(-(ac2 + P.bc[2])));
    P.out[g] = density;
    P.out[P.n + g * 3 + 0] = c0;
    P.out[P.n + g * 3 + 1] = c1;
    P.out[P.n + g * 3 + 2] = c2;
  }
}

// ------------------------------ launcher -----------------------------------
extern "C" void kernel_launch(void* const* d_in, const int* in_sizes, int n_in,
                              void* d_out, int out_size, void* d_ws, size_t ws_size,
                              hipStream_t stream) {
  const float* pos  = (const float*)d_in[0];
  const float* view = (const float*)d_in[1];
  const float* w0   = (const float*)d_in[10];
  const float* b0   = (const float*)d_in[11];
  const float* wh   = (const float*)d_in[12];
  const float* bh   = (const float*)d_in[13];
  const float* wd   = (const float*)d_in[14];
  const float* bd   = (const float*)d_in[15];
  const float* wc   = (const float*)d_in[16];
  const float* bc   = (const float*)d_in[17];

  half_t* wt = (half_t*)d_ws;  // needs (24576 + 458752)*2 = ~0.95 MB scratch
  const int n = in_sizes[0] / 3;

  const int prep_total  = WT0_ELEMS + WTH_ELEMS;
  const int prep_blocks = (prep_total + 255) / 256;
  nerf_prep_weights<<<prep_blocks, 256, 0, stream>>>(w0, wh, wt);

  NerfParams P;
  P.pos = pos; P.view = view;
  for (int l = 0; l < NLVL; ++l) P.tab[l] = (const float*)d_in[2 + l];
  P.b0 = b0; P.bh = bh;
  P.wd = wd; P.bd = bd; P.wc = wc; P.bc = bc;
  P.wt0 = wt;
  P.wth = wt + WT0_ELEMS;
  P.out = (float*)d_out;
  P.n = n;

  nerf_main<<<dim3(n / ROWS), dim3(THREADS), 0, stream>>>(P);
}
